// ContextVectorLLM_45603962749567
// MI455X (gfx1250) — matrix operand order repaired and checked by reading, not compiled, not run on hardware
//
#include <hip/hip_runtime.h>
#include <hip/hip_bf16.h>

#define VOCAB 32000
#define EMBED 512
#define CTXD  256
#define HID   1024
#define NB    4
#define SEQ   1024
#define K0    (EMBED + CTXD)   // 768

#define TM 128
#define TN 128
#define TK 128
#define NKB (HID / TK)         // 8
#define ABUF_BYTES (TM * TK * 2)  // 32768
#define STAGE_BYTES (2 * ABUF_BYTES)          // A + B per stage = 64KB
#define LDS_BYTES (2 * STAGE_BYTES)           // double buffered = 128KB

typedef __attribute__((ext_vector_type(16))) __bf16 v16bf;
typedef __attribute__((ext_vector_type(8)))  float  v8f;
typedef __attribute__((ext_vector_type(4)))  float  v4f;
typedef __attribute__((ext_vector_type(4)))  unsigned int u32x4;
typedef __attribute__((ext_vector_type(8)))  int    i32x8;
typedef __attribute__((ext_vector_type(4)))  int    i32x4;

union Frag { v16bf bf; v4f f4[2]; };

// ---------------------------------------------------------------------------
// Tensor Data Mover: build a 2D-tile D# per CDNA5 ISA and
// issue tensor_load_to_lds. data_size = 2 bytes (bf16). Tile rows contiguous
// into LDS ([tile_d1][tile_d0] row-major).
// ---------------------------------------------------------------------------
static __device__ __forceinline__ void tdm_load_tile_bf16(
    const void* gtile, unsigned lds_addr,
    unsigned tensor_d0, unsigned tensor_d1, unsigned long long stride_d0,
    unsigned tile_d0, unsigned tile_d1)
{
  unsigned long long ga = (unsigned long long)gtile;
  u32x4 g0;
  g0.x = 1u;                                   // count=1, no gather
  g0.y = lds_addr;                             // LDS byte address
  g0.z = (unsigned)(ga & 0xffffffffull);       // global_addr[31:0]
  g0.w = (unsigned)((ga >> 32) & 0x1ffffffull) // global_addr[56:32]
         | (2u << 30);                         // type = 2 ("image")
  i32x8 g1;
  g1.s0 = (int)(1u << 16);                     // data_size=1 -> 2 bytes; mask=0
  g1.s1 = (int)((tensor_d0 & 0xffffu) << 16);                         // dim0[15:0]
  g1.s2 = (int)(((tensor_d0 >> 16) & 0xffffu) | ((tensor_d1 & 0xffffu) << 16));
  g1.s3 = (int)(((tensor_d1 >> 16) & 0xffffu) | ((tile_d0 & 0xffffu) << 16));
  g1.s4 = (int)(tile_d1 & 0xffffu);            // tile_dim1, tile_dim2=0
  g1.s5 = (int)(stride_d0 & 0xffffffffull);    // dim0 stride low
  g1.s6 = (int)((stride_d0 >> 32) & 0xffffull);
  g1.s7 = 0;
  i32x4 z = {0, 0, 0, 0};
#if __has_include(<hip/amd_detail/amd_gfx1250_TDM.h>)
  i32x8 z8 = {0, 0, 0, 0, 0, 0, 0, 0};
  __builtin_amdgcn_tensor_load_to_lds(g0, g1, z, z, z8, 0);
#else
  __builtin_amdgcn_tensor_load_to_lds(g0, g1, z, z, 0);
#endif
}

// ---------------------------------------------------------------------------
// Kernel 1: embeddings  x[b,s,:] = tok_emb[ids[b,s]] + pos_emb[s]
// ---------------------------------------------------------------------------
__global__ void k_embed(const int* __restrict__ ids,
                        const float* __restrict__ tok,
                        const float* __restrict__ pos,
                        float* __restrict__ X)
{
  int bs = blockIdx.x;               // 0 .. NB*SEQ-1
  int e  = threadIdx.x;              // 0 .. EMBED-1
  int s  = bs & (SEQ - 1);
  int id = ids[bs];
  X[(size_t)bs * EMBED + e] = tok[(size_t)id * EMBED + e] + pos[(size_t)s * EMBED + e];
}

// ---------------------------------------------------------------------------
// Kernel 2: f32 -> bf16 conversion (Wout)
// ---------------------------------------------------------------------------
__global__ void k_cvt_bf16(const float* __restrict__ src,
                           __hip_bfloat16* __restrict__ dst, int n)
{
  int i = blockIdx.x * blockDim.x + threadIdx.x;
  if (i < n) dst[i] = __float2bfloat16(src[i]);
}

// ---------------------------------------------------------------------------
// Kernel 3: sequential ctx recurrence. One workgroup per batch chain,
// 1024 threads (= 32 wave32). ctx/h live in LDS; weights stream from L2.
// Emits traj (ctx before update) and H (bf16) for the batched logits GEMM.
// ---------------------------------------------------------------------------
__global__ void __launch_bounds__(1024)
k_recur(const float* __restrict__ X,
        const float* __restrict__ W0, const float* __restrict__ b0,
        const float* __restrict__ W1, const float* __restrict__ b1,
        const float* __restrict__ Wc, const float* __restrict__ bc,
        __hip_bfloat16* __restrict__ Hbf, float* __restrict__ traj)
{
  __shared__ float xin[K0];
  __shared__ float h0s[HID];
  __shared__ float h1s[HID];
  __shared__ float ctx[CTXD];

  int b = blockIdx.x, tid = threadIdx.x;
  if (tid < CTXD) ctx[tid] = 0.f;

  const v4f* w0r = (const v4f*)(W0 + (size_t)tid * K0);
  const v4f* w1r = (const v4f*)(W1 + (size_t)tid * HID);
  const v4f* wcr = (const v4f*)(Wc + (size_t)(tid < CTXD ? tid : 0) * HID);
  float bb0 = b0[tid], bb1 = b1[tid];
  float bbc = (tid < CTXD) ? bc[tid] : 0.f;
  const float* xb = X + (size_t)b * SEQ * EMBED;
  __syncthreads();

  for (int s = 0; s < SEQ; ++s) {
    if (tid < EMBED)      xin[tid] = xb[(size_t)s * EMBED + tid];
    else if (tid < K0)    xin[tid] = ctx[tid - EMBED];
    if (tid < CTXD)       traj[((size_t)b * SEQ + s) * CTXD + tid] = ctx[tid];
    __syncthreads();

    // layer 0: h0 = relu(W0 @ [x; ctx] + b0)
    {
      v4f a = {0.f, 0.f, 0.f, 0.f};
      const v4f* xv = (const v4f*)xin;
      #pragma unroll 4
      for (int k = 0; k < K0 / 4; ++k) a += w0r[k] * xv[k];
      float h = a.x + a.y + a.z + a.w + bb0;
      h0s[tid] = h > 0.f ? h : 0.f;
    }
    __syncthreads();

    // layer 1: h1 = relu(W1 @ h0 + b1)
    {
      v4f a = {0.f, 0.f, 0.f, 0.f};
      const v4f* hv = (const v4f*)h0s;
      #pragma unroll 4
      for (int k = 0; k < HID / 4; ++k) a += w1r[k] * hv[k];
      float h = a.x + a.y + a.z + a.w + bb1;
      h = h > 0.f ? h : 0.f;
      h1s[tid] = h;
      Hbf[((size_t)b * SEQ + s) * HID + tid] = __float2bfloat16(h);
    }
    __syncthreads();

    // ctx update: ctx += Wc @ h1 + bc
    if (tid < CTXD) {
      v4f a = {0.f, 0.f, 0.f, 0.f};
      const v4f* hv = (const v4f*)h1s;
      #pragma unroll 4
      for (int k = 0; k < HID / 4; ++k) a += wcr[k] * hv[k];
      ctx[tid] += a.x + a.y + a.z + a.w + bbc;
    }
    __syncthreads();
  }
}

// ---------------------------------------------------------------------------
// Kernel 4: logits[4096, 32000] = H(bf16) @ Wout(bf16)^T + bout, fp32 acc.
// 256 threads (8 waves). Block tile 128M x 128N, K staged in 128-wide blocks
// via TDM into double-buffered LDS; each wave owns 16 M-rows x full 128 N
// (8 v8f accumulators), v_wmma_f32_16x16x32_bf16 inner loop.
// ---------------------------------------------------------------------------
__global__ void __launch_bounds__(256)
k_logits(const __hip_bfloat16* __restrict__ Hbf,
         const __hip_bfloat16* __restrict__ Wbf,
         const float* __restrict__ bout,
         float* __restrict__ lg)
{
  extern __shared__ char smem[];
  unsigned lds_base = (unsigned)(unsigned long long)(void*)smem;

  int n0 = blockIdx.x * TN;
  int m0 = blockIdx.y * TM;
  int tid  = threadIdx.x;
  int wave = tid >> 5;
  int lane = tid & 31;
  int l16  = lane & 15;
  int hl   = lane >> 4;     // lane half: 0 or 1

  v8f acc[8];
  v8f zero = {0.f, 0.f, 0.f, 0.f, 0.f, 0.f, 0.f, 0.f};
  #pragma unroll
  for (int i = 0; i < 8; ++i) acc[i] = zero;

  // prologue: stage k-block 0 into buffer 0 (wave 0 only; TDM ignores EXEC)
  if (tid < 32) {
    tdm_load_tile_bf16(Hbf + (size_t)m0 * HID, lds_base,
                       HID, NB * SEQ, HID, TK, TM);
    tdm_load_tile_bf16(Wbf + (size_t)n0 * HID, lds_base + ABUF_BYTES,
                       HID, VOCAB, HID, TK, TN);
  }

  int buf = 0;
  for (int kb = 0; kb < NKB; ++kb) {
    if (tid < 32) {
      if (kb + 1 < NKB) {
        unsigned nb = lds_base + (buf ^ 1) * STAGE_BYTES;
        tdm_load_tile_bf16(Hbf + (size_t)m0 * HID + (kb + 1) * TK, nb,
                           HID, NB * SEQ, HID, TK, TM);
        tdm_load_tile_bf16(Wbf + (size_t)n0 * HID + (kb + 1) * TK, nb + ABUF_BYTES,
                           HID, VOCAB, HID, TK, TN);
        __builtin_amdgcn_s_wait_tensorcnt(2);   // current stage's 2 loads done
      } else {
        __builtin_amdgcn_s_wait_tensorcnt(0);
      }
    }
    __syncthreads();

    const char* aT = smem + buf * STAGE_BYTES;               // [TM][TK] bf16
    const char* bT = smem + buf * STAGE_BYTES + ABUF_BYTES;  // [TN][TK] bf16
    const v4f* arow = (const v4f*)(aT + (size_t)(wave * 16 + l16) * (TK * 2));

    #pragma unroll
    for (int ks = 0; ks < TK / 32; ++ks) {
      Frag af;
      // 16-bit A layout: lanes<16 hold K 0-7 & 16-23, lanes>=16 hold 8-15 & 24-31
      af.f4[0] = arow[ks * 4 + hl];
      af.f4[1] = arow[ks * 4 + hl + 2];
      #pragma unroll
      for (int ns = 0; ns < 8; ++ns) {
        const v4f* brow = (const v4f*)(bT + (size_t)(ns * 16 + l16) * (TK * 2));
        Frag bfr;
        // B layout: lanes<16 hold K 0-7 & 16-23 of column N=l16, lanes>=16 hold K 8-15 & 24-31
        bfr.f4[0] = brow[ks * 4 + hl];
        bfr.f4[1] = brow[ks * 4 + hl + 2];
        acc[ns] = __builtin_amdgcn_wmma_f32_16x16x32_bf16(
            false, af.bf, false, bfr.bf, (short)0, acc[ns], false, false);
      }
    }
    __syncthreads();   // all waves done reading before next TDM reuses buffer
    buf ^= 1;
  }

  // epilogue: add bias, store. C layout: VGPR r -> M = r + 8*half, N = l16.
  #pragma unroll
  for (int ns = 0; ns < 8; ++ns) {
    int n = n0 + ns * 16 + l16;
    float bias = bout[n];
    size_t rowbase = (size_t)(m0 + wave * 16 + hl * 8) * VOCAB + (size_t)n;
    #pragma unroll
    for (int r = 0; r < 8; ++r)
      lg[rowbase + (size_t)r * VOCAB] = acc[ns][r] + bias;
  }
}

// ---------------------------------------------------------------------------
extern "C" void kernel_launch(void* const* d_in, const int* in_sizes, int n_in,
                              void* d_out, int out_size, void* d_ws, size_t ws_size,
                              hipStream_t stream)
{
  const int*   ids  = (const int*)d_in[0];
  const float* tok  = (const float*)d_in[1];
  const float* pos  = (const float*)d_in[2];
  const float* W0   = (const float*)d_in[3];
  const float* b0   = (const float*)d_in[4];
  const float* W1   = (const float*)d_in[5];
  const float* b1   = (const float*)d_in[6];
  const float* Wout = (const float*)d_in[7];
  const float* bout = (const float*)d_in[8];
  const float* Wc   = (const float*)d_in[9];
  const float* bc   = (const float*)d_in[10];

  float* lg   = (float*)d_out;                        // [NB, SEQ, VOCAB]
  float* traj = lg + (size_t)NB * SEQ * VOCAB;        // [NB, SEQ, CTXD]

  char* ws = (char*)d_ws;
  size_t xBytes = (size_t)NB * SEQ * EMBED * sizeof(float);   // 8 MB
  size_t hBytes = (size_t)NB * SEQ * HID * 2;                 // 8 MB
  float*          X    = (float*)ws;
  __hip_bfloat16* Hbf  = (__hip_bfloat16*)(ws + xBytes);
  __hip_bfloat16* Wbf  = (__hip_bfloat16*)(ws + xBytes + hBytes);  // 65.5 MB

  (void)in_sizes; (void)n_in; (void)out_size; (void)ws_size;

  hipFuncSetAttribute(reinterpret_cast<const void*>(k_logits),
                      hipFuncAttributeMaxDynamicSharedMemorySize, LDS_BYTES);

  k_embed<<<NB * SEQ, EMBED, 0, stream>>>(ids, tok, pos, X);

  int nW = VOCAB * HID;
  k_cvt_bf16<<<(nW + 255) / 256, 256, 0, stream>>>(Wout, Wbf, nW);

  k_recur<<<NB, HID, 0, stream>>>(X, W0, b0, W1, b1, Wc, bc, Hbf, traj);

  dim3 g(VOCAB / TN, (NB * SEQ) / TM);
  k_logits<<<g, 256, LDS_BYTES, stream>>>(Hbf, Wbf, bout, lg);
}
